// BasicModelOneTheta_67645734912718
// MI455X (gfx1250) — matrix operand order repaired and checked by reading, not compiled, not run on hardware
//
#include <hip/hip_runtime.h>
#include <hip/hip_bf16.h>

// ---------------- problem constants ----------------
constexpr int T  = 256;
constexpr int Bn = 256;
constexpr int D  = 128;
constexpr int H  = 512;
constexpr int Pn = 3;
constexpr int DX = D + 1;        // 129
constexpr int H1 = H + 1;        // 513
constexpr int K1 = 10 * D;       // 1280
constexpr int KXP = 160;         // x K padded (129 -> 160)
constexpr int KXPC = 5;          // x K chunks
constexpr int KHC  = 16;         // h K chunks (512/32)
constexpr int KW1C = 16;         // w1 main K chunks (512/32)
constexpr int KW2C = 40;         // w2 K chunks (1280/32)
constexpr int PRE_STRIDE = 1288; // 1280 + 8 pad (bf16 elems) -> conflict-free LDS

// ---------------- WMMA types ----------------
typedef __attribute__((ext_vector_type(16))) __bf16 v16bf;
typedef __attribute__((ext_vector_type(8)))  float  v8f;

union BF16x16 { v16bf v; unsigned u[8]; uint4 q[2]; };

// f32 -> bf16 round-to-nearest-even (used only in prep kernels / epilogues)
static __device__ inline unsigned short f2bf(float f) {
  unsigned u = __float_as_uint(f);
  u += 0x7FFFu + ((u >> 16) & 1u);
  return (unsigned short)(u >> 16);
}
static __device__ inline unsigned pk2(float a, float b) {
  return (unsigned)f2bf(a) | ((unsigned)f2bf(b) << 16);
}

static __device__ inline void load_panel(BF16x16& b, const unsigned char* base,
                                         int panel, int lane) {
  const uint4* p = (const uint4*)(base + (size_t)panel * 1024 + lane * 32);
  b.q[0] = p[0];
  b.q[1] = p[1];
}

// A operand: lane owns row M=lane&15, K segments [ks, ks+8) and [ks+16, ks+24),
// ks = kc*32 + (lane>>4)*8. With bf16 row-major rows this is two 16B loads.
static __device__ inline void load_a_bf16(BF16x16& a, const uint4* row4, int ks) {
  a.q[0] = row4[ks >> 3];
  a.q[1] = row4[(ks >> 3) + 2];
}

static __device__ inline v8f wmma_bf16(const BF16x16& a, const BF16x16& b, v8f c) {
  return __builtin_amdgcn_wmma_f32_16x16x32_bf16(
      /*neg_a=*/false, a.v, /*neg_b=*/false, b.v,
      /*c_mod=*/(short)0, c, /*reuse_a=*/false, /*reuse_b=*/false);
}

static __device__ inline float sigm(float x) { return 1.0f / (1.0f + __expf(-x)); }

// ---------------- prep: f32 rows -> zero-padded bf16 rows ----------------
__global__ void cvt_bf16_rows(const float* __restrict__ src, unsigned* __restrict__ dst,
                              long long nPairs, int Ksrc, int KdstPairs) {
  long long i = (long long)blockIdx.x * blockDim.x + threadIdx.x;
  if (i >= nPairs) return;
  long long row = i / KdstPairs;
  int pr = (int)(i - row * KdstPairs);
  int k0 = 2 * pr, k1 = 2 * pr + 1;
  const float* s = src + row * Ksrc;
  float a = (k0 < Ksrc) ? s[k0] : 0.f;
  float b = (k1 < Ksrc) ? s[k1] : 0.f;
  dst[i] = pk2(a, b);
}

// ---------------- prep: weights -> WMMA B-panel layout ----------------
// One 32-thread wave packs one 1KB panel = 16x32 bf16 tile of W^T.
// Lane L holds column n = nt*16 + (L&15), K = kc*32 + (L>>4)*8 + [0,8) and +16 + [0,8).
__global__ void pack_b_panels(const float* __restrict__ src, unsigned char* __restrict__ dst,
                              int N, int Ksrc, int rowStride, int KC) {
  const int panel = blockIdx.x;
  const int kc = panel % KC, nt = panel / KC;
  const int lane = threadIdx.x & 31;
  const int n = nt * 16 + (lane & 15);
  const int kb = kc * 32 + (lane >> 4) * 8;
  unsigned short tmp[16];
#pragma unroll
  for (int i = 0; i < 16; ++i) {
    int k = kb + (i < 8 ? i : i + 8);
    float v = (n < N && k < Ksrc) ? src[(size_t)n * rowStride + k] : 0.0f;
    tmp[i] = f2bf(v);
  }
  unsigned* out = (unsigned*)(dst + (size_t)panel * 1024 + lane * 32);
#pragma unroll
  for (int i = 0; i < 8; ++i)
    out[i] = (unsigned)tmp[2 * i] | ((unsigned)tmp[2 * i + 1] << 16);
}

// ---------------- GRU step: one wave per 16x16 h-tile ----------------
// Inner loops are pure b128 loads + v_wmma: A operands come pre-converted
// (xbf for inputs, hbf ping-pong for the carried state).
__global__ __launch_bounds__(256) void gru_step(
    const unsigned* __restrict__ xbf,          // (T,B,KXP) bf16 as uint pairs
    const int* __restrict__ lengths,
    const float* __restrict__ b_ih, const float* __restrict__ b_hh,
    const unsigned char* __restrict__ pIH, const unsigned char* __restrict__ pHH,
    const float* __restrict__ hin, const unsigned short* __restrict__ hbf_in,
    float* __restrict__ hout, unsigned short* __restrict__ hbf_out,
    unsigned short* __restrict__ hiddenOut, int t) {
  const int lane = threadIdx.x & 31;
  const int wave = blockIdx.x * 8 + (threadIdx.x >> 5);   // 0..511
  const int bi = wave >> 5, ji = wave & 31;               // 16 b-tiles x 32 j-tiles
  const int bbase = bi * 16, jbase = ji * 16;
  const int m = lane & 15, hi4 = lane >> 4;

  v8f accR = {}, accZ = {}, accXN = {}, accHN = {};

  // ---- phase 1: x_t @ w_ih^T (K padded to 160, zeros baked into xbf/panels) ----
  const uint4* xrow4 = (const uint4*)(xbf + ((size_t)t * Bn + (bbase + m)) * (KXP / 2));
#pragma unroll
  for (int kc = 0; kc < KXPC; ++kc) {
    const int ks = kc * 32 + hi4 * 8;
    BF16x16 a; load_a_bf16(a, xrow4, ks);
    BF16x16 br_, bz_, bn_;
    load_panel(br_, pIH, (0 * 32 + ji) * KXPC + kc, lane);
    load_panel(bz_, pIH, (1 * 32 + ji) * KXPC + kc, lane);
    load_panel(bn_, pIH, (2 * 32 + ji) * KXPC + kc, lane);
    accR  = wmma_bf16(a, br_, accR);
    accZ  = wmma_bf16(a, bz_, accZ);
    accXN = wmma_bf16(a, bn_, accXN);
  }

  // ---- phase 2: h_t @ w_hh^T (K = 512), A from bf16 carried state ----
  const uint4* hrow4 = (const uint4*)(hbf_in + (size_t)(bbase + m) * H);
#pragma unroll 4
  for (int kc = 0; kc < KHC; ++kc) {
    const int ks = kc * 32 + hi4 * 8;
    BF16x16 a; load_a_bf16(a, hrow4, ks);
    BF16x16 br_, bz_, bn_;
    load_panel(br_, pHH, (0 * 32 + ji) * KHC + kc, lane);
    load_panel(bz_, pHH, (1 * 32 + ji) * KHC + kc, lane);
    load_panel(bn_, pHH, (2 * 32 + ji) * KHC + kc, lane);
    accR  = wmma_bf16(a, br_, accR);
    accZ  = wmma_bf16(a, bz_, accZ);
    accHN = wmma_bf16(a, bn_, accHN);
  }

  // ---- gate epilogue (C layout: element(M,N): M = v + 8*hi, N = lane&15) ----
  const int nn = lane & 15;
  const int col = jbase + nn;
  const float br  = b_ih[col] + b_hh[col];
  const float bz  = b_ih[H + col] + b_hh[H + col];
  const float bxn = b_ih[2 * H + col];
  const float bhn = b_hh[2 * H + col];
#pragma unroll
  for (int v = 0; v < 8; ++v) {
    const int row = bbase + v + 8 * hi4;
    const float hold = hin[(size_t)row * H + col];
    const float r  = sigm(accR[v] + br);
    const float z  = sigm(accZ[v] + bz);
    const float nv = tanhf(accXN[v] + bxn + r * (accHN[v] + bhn));
    const float hnew = (1.f - z) * nv + z * hold;
    const bool valid = t < lengths[row];
    const float hcar = valid ? hnew : hold;
    hout[(size_t)row * H + col] = hcar;                                // f32 carry
    hbf_out[(size_t)row * H + col] = f2bf(hcar);                       // bf16 A copy
    hiddenOut[((size_t)row * T + t) * H + col] = f2bf(valid ? hnew : 0.f); // outs
  }
}

// ---------------- fused feedforward: pre = sigmoid(hwt@w1^T+b1); preds = pre@w2^T+b2 ----
__global__ __launch_bounds__(256) void ffn_kernel(
    const unsigned short* __restrict__ hidden, const float* __restrict__ x,
    const int* __restrict__ lengths,
    const float* __restrict__ w1, const float* __restrict__ b1,
    const unsigned char* __restrict__ pW1, const unsigned char* __restrict__ pW2,
    const float* __restrict__ b2, float* __restrict__ preds) {
  __shared__ __align__(16) unsigned short preS[16 * PRE_STRIDE]; // 41.2 KB of 320 KB WGP LDS
  __shared__ float tdsh[16];

  const int b  = blockIdx.x >> 4;
  const int t0 = (blockIdx.x & 15) * 16;
  const int lane = threadIdx.x & 31;
  const int w = threadIdx.x >> 5;       // 8 waves
  const int m = lane & 15, hi4 = lane >> 4;
  const int nn = lane & 15;

  if (threadIdx.x < 16) {               // time deltas for the rank-1 (H+1)-th column
    int tt = t0 + threadIdx.x;
    float td = 0.f;
    if (tt < T - 1)
      td = x[((size_t)(tt + 1) * Bn + b) * DX] - x[((size_t)tt * Bn + b) * DX];
    tdsh[threadIdx.x] = td;
  }
  __syncthreads();

  // Register-cache all 16 A chunks of this wave's 16 hidden rows (128 VGPRs).
  const uint4* hrow4 = (const uint4*)(hidden + ((size_t)b * T + (t0 + m)) * H);
  BF16x16 aCh[KW1C];
#pragma unroll
  for (int kc = 0; kc < KW1C; ++kc)
    load_a_bf16(aCh[kc], hrow4, kc * 32 + hi4 * 8);

  // ---- stage 1: 16 x 1280 pre tile (each wave: 10 n-tiles) ----
  for (int ntl = 0; ntl < 10; ++ntl) {
    const int nt = w * 10 + ntl;
    v8f acc = {};
#pragma unroll 4
    for (int kc = 0; kc < KW1C; ++kc) {
      BF16x16 bb; load_panel(bb, pW1, nt * KW1C + kc, lane);
      acc = wmma_bf16(aCh[kc], bb, acc);
    }
    const int col = nt * 16 + nn;
    const float bias  = b1[col];
    const float wlast = w1[(size_t)col * H1 + H];  // w1[:,512] (td column), kept f32
#pragma unroll
    for (int v = 0; v < 8; ++v) {
      float pre = acc[v] + bias + tdsh[v + 8 * hi4] * wlast;
      preS[(size_t)(v + 8 * hi4) * PRE_STRIDE + col] = f2bf(sigm(pre));
    }
  }
  __syncthreads();

  // ---- stage 2: 16 x 128 preds tile (each wave: one 16-col tile, K=1280 from LDS) ----
  {
    v8f acc = {};
    const unsigned* rp = (const unsigned*)&preS[(size_t)m * PRE_STRIDE];
#pragma unroll 4
    for (int kc = 0; kc < KW2C; ++kc) {
      const int ks = kc * 32 + hi4 * 8;
      BF16x16 a;
#pragma unroll
      for (int i = 0; i < 4; ++i) { a.u[i] = rp[(ks >> 1) + i]; a.u[4 + i] = rp[((ks + 16) >> 1) + i]; }
      BF16x16 bb; load_panel(bb, pW2, w * KW2C + kc, lane);
      acc = wmma_bf16(a, bb, acc);
    }
    const int d = w * 16 + nn;
    const float bias = b2[d];
    const int lim = lengths[b] - 1;
#pragma unroll
    for (int v = 0; v < 8; ++v) {
      const int tt = t0 + v + 8 * hi4;
      if (tt < T - 1) {
        float val = (tt < lim) ? (acc[v] + bias) : 0.f;   // reference mask
        preds[((size_t)b * (T - 1) + tt) * D + d] = val;
      }
    }
  }
}

// ---------------- tail: dist_params = exp(-(h_last @ wp^T + bp)); emit lengths ----
__global__ void tail_kernel(const float* __restrict__ hlast, const float* __restrict__ wp,
                            const float* __restrict__ bp, const int* __restrict__ lengths,
                            float* __restrict__ out) {
  const int b = threadIdx.x;
  if (b < Bn) {
    const float* h = hlast + (size_t)b * H;
    float s0 = 0.f, s1 = 0.f, s2 = 0.f;
    for (int k = 0; k < H; ++k) {
      float hv = h[k];
      s0 += hv * wp[k]; s1 += hv * wp[H + k]; s2 += hv * wp[2 * H + k];
    }
    out[b * Pn + 0] = __expf(-(s0 + bp[0]));
    out[b * Pn + 1] = __expf(-(s1 + bp[1]));
    out[b * Pn + 2] = __expf(-(s2 + bp[2]));
    out[Bn * Pn + b] = (float)lengths[b];   // third tuple output
  }
}

// ---------------- launch ----------------
extern "C" void kernel_launch(void* const* d_in, const int* in_sizes, int n_in,
                              void* d_out, int out_size, void* d_ws, size_t ws_size,
                              hipStream_t stream) {
  (void)in_sizes; (void)n_in; (void)out_size; (void)ws_size;
  const float* x    = (const float*)d_in[0];
  const float* h0   = (const float*)d_in[1];
  const int*  lens  = (const int*)d_in[2];
  const float* w_ih = (const float*)d_in[3];
  const float* w_hh = (const float*)d_in[4];
  const float* b_ih = (const float*)d_in[5];
  const float* b_hh = (const float*)d_in[6];
  const float* w1   = (const float*)d_in[7];
  const float* b1   = (const float*)d_in[8];
  const float* w2   = (const float*)d_in[9];
  const float* b2   = (const float*)d_in[10];
  const float* wp   = (const float*)d_in[11];
  const float* bp   = (const float*)d_in[12];

  unsigned char* wsb = (unsigned char*)d_ws;
  size_t o = 0;
  auto carve = [&](size_t bytes) { size_t r = o; o += (bytes + 255) & ~(size_t)255; return r; };
  unsigned char* pIH = wsb + carve((size_t)96 * KXPC * 1024);  //  480 KB
  unsigned char* pHH = wsb + carve((size_t)96 * KHC  * 1024);  // 1.5 MB
  unsigned char* pW1 = wsb + carve((size_t)80 * KW1C * 1024);  // 1.25 MB
  unsigned char* pW2 = wsb + carve((size_t)8  * KW2C * 1024);  //  320 KB
  float* hbuf0 = (float*)(wsb + carve((size_t)Bn * H * 4));    //  512 KB f32 ping
  float* hbuf1 = (float*)(wsb + carve((size_t)Bn * H * 4));    //  512 KB f32 pong
  unsigned short* hbf0 = (unsigned short*)(wsb + carve((size_t)Bn * H * 2)); // bf16 ping
  unsigned short* hbf1 = (unsigned short*)(wsb + carve((size_t)Bn * H * 2)); // bf16 pong
  unsigned* xbf = (unsigned*)(wsb + carve((size_t)T * Bn * KXP * 2));        // 21 MB
  unsigned short* hidden = (unsigned short*)(wsb + carve((size_t)Bn * T * H * 2)); // 64 MB

  // 1) pack weights into WMMA B-panel layout + pre-convert activations (parallel)
  pack_b_panels<<<96 * KXPC, 32, 0, stream>>>(w_ih, pIH, 3 * H, DX, DX, KXPC);
  pack_b_panels<<<96 * KHC,  32, 0, stream>>>(w_hh, pHH, 3 * H, H,  H,  KHC);
  pack_b_panels<<<80 * KW1C, 32, 0, stream>>>(w1,   pW1, K1,    H,  H1, KW1C);
  pack_b_panels<<<8  * KW2C, 32, 0, stream>>>(w2,   pW2, D,     K1, K1, KW2C);
  {
    long long nP = (long long)T * Bn * (KXP / 2);  // x -> xbf, zero-padded K 129->160
    cvt_bf16_rows<<<(int)((nP + 255) / 256), 256, 0, stream>>>(x, xbf, nP, DX, KXP / 2);
  }
  {
    long long nP = (long long)Bn * (H / 2);        // h0 -> bf16 ping
    cvt_bf16_rows<<<(int)((nP + 255) / 256), 256, 0, stream>>>(h0, (unsigned*)hbf0, nP, H, H / 2);
  }
  hipMemcpyAsync(hbuf0, h0, (size_t)Bn * H * 4, hipMemcpyDeviceToDevice, stream);

  // 2) 256 sequential GRU steps (ping-pong; final state lands in buffers 0)
  for (int t = 0; t < T; ++t) {
    float* hin  = (t & 1) ? hbuf1 : hbuf0;
    float* hout = (t & 1) ? hbuf0 : hbuf1;
    unsigned short* hbin  = (t & 1) ? hbf1 : hbf0;
    unsigned short* hbout = (t & 1) ? hbf0 : hbf1;
    gru_step<<<64, 256, 0, stream>>>(xbf, lens, b_ih, b_hh, pIH, pHH,
                                     hin, hbin, hout, hbout, hidden, t);
  }

  // 3) fused two-layer feedforward -> preds (B, T-1, D)
  ffn_kernel<<<Bn * 16, 256, 0, stream>>>(hidden, x, lens, w1, b1, pW1, pW2, b2,
                                          (float*)d_out);

  // 4) dist_params + lengths appended after preds
  tail_kernel<<<1, 256, 0, stream>>>(hbuf0, wp, bp, lens,
                                     (float*)d_out + (size_t)Bn * (T - 1) * D);
}
